// MoELayer_77455440216222
// MI455X (gfx1250) — compile-verified, matrix operand order checked by reading, not run on hardware
//
#include <hip/hip_runtime.h>
#include <hip/hip_bf16.h>
#include <stdint.h>

// Problem constants (match reference)
static constexpr int N = 8192;
static constexpr int D = 1024;
static constexpr int F = 4096;
static constexpr int E = 8;

typedef __bf16 bf16;
typedef __attribute__((ext_vector_type(16))) __bf16 v16bf;
typedef __attribute__((ext_vector_type(4)))  __bf16 v4bf;
typedef __attribute__((ext_vector_type(8)))  float  v8f;

__device__ __forceinline__ v8f wmma_bf16(v16bf a, v16bf b, v8f c) {
  // 8 args: (neg_a, A, neg_b, B, c_mod, C, reuse_a, reuse_b)
  return __builtin_amdgcn_wmma_f32_16x16x32_bf16(false, a, false, b,
                                                 (short)0, c, false, false);
}

// fast sigmoid: v_exp_f32 + v_add + v_rcp_f32 (avoids IEEE div_scale sequence)
__device__ __forceinline__ float fast_sigmoid(float v) {
  return __builtin_amdgcn_rcpf(1.f + __expf(-v));
}

// ---------------------------------------------------------------------------
// Kernel 1: gating. One wave32 per token: logits = x . Wg + bg, top-2,
// 2-way softmax, append (token, weight) to per-expert compacted lists.
// ---------------------------------------------------------------------------
__global__ __launch_bounds__(256)
void moe_gate(const float* __restrict__ x, const float* __restrict__ Wg,
              const float* __restrict__ bg, int* __restrict__ counts,
              int* __restrict__ tok_ids, float* __restrict__ tok_wts) {
  const int lane = threadIdx.x & 31;
  const int n = blockIdx.x * 8 + (threadIdx.x >> 5);   // token id
  if (n >= N) return;

  float acc[E];
#pragma unroll
  for (int e = 0; e < E; ++e) acc[e] = 0.f;

  const float* xr = x + (size_t)n * D;
  for (int d = lane; d < D; d += 32) {
    const float xv = xr[d];
    const float* wr = Wg + (size_t)d * E;
#pragma unroll
    for (int e = 0; e < E; ++e) acc[e] += xv * wr[e];
  }
#pragma unroll
  for (int e = 0; e < E; ++e) {
#pragma unroll
    for (int off = 16; off > 0; off >>= 1)
      acc[e] += __shfl_xor(acc[e], off, 32);
  }
  if (lane == 0) {
#pragma unroll
    for (int e = 0; e < E; ++e) acc[e] += bg[e];
    int e0 = 0;
#pragma unroll
    for (int e = 1; e < E; ++e) if (acc[e] > acc[e0]) e0 = e;
    int e1 = (e0 == 0) ? 1 : 0;
#pragma unroll
    for (int e = 0; e < E; ++e) if (e != e0 && acc[e] > acc[e1]) e1 = e;
    const float p1 = __expf(acc[e1] - acc[e0]);     // softmax over the 2 picks
    const float inv = __builtin_amdgcn_rcpf(1.f + p1);
    int p = atomicAdd(&counts[e0], 1);
    tok_ids[e0 * N + p] = n; tok_wts[e0 * N + p] = inv;
    p = atomicAdd(&counts[e1], 1);
    tok_ids[e1 * N + p] = n; tok_wts[e1 * N + p] = p1 * inv;
  }
}

// ---------------------------------------------------------------------------
// Kernel 2: inputs fp32 -> bf16 (keeps HBM traffic for X at 16 MB)
// ---------------------------------------------------------------------------
__global__ __launch_bounds__(256)
void cvt_x_bf16(const float* __restrict__ in, bf16* __restrict__ out) {
  const size_t i = (size_t)blockIdx.x * blockDim.x + threadIdx.x;
  const float4 v = ((const float4*)in)[i];
  v4bf o; o[0] = (bf16)v.x; o[1] = (bf16)v.y; o[2] = (bf16)v.z; o[3] = (bf16)v.w;
  ((v4bf*)out)[i] = o;
}

// ---------------------------------------------------------------------------
// Kernel 3: per-expert fp32 [R][C] -> bf16 transposed [C][R] via LDS tile.
// Transposed bf16 weights (128 MB total) stay resident in the 192 MB L2.
// ---------------------------------------------------------------------------
__global__ __launch_bounds__(256)
void transpose_bf16(const float* __restrict__ in, bf16* __restrict__ out,
                    int R, int C) {
  __shared__ float t[32][33];
  const size_t eoff = (size_t)blockIdx.z * R * C;
  in  += eoff; out += eoff;
  const int c0 = blockIdx.x * 32, r0 = blockIdx.y * 32;
#pragma unroll
  for (int i = 0; i < 4; ++i) {
    const int r = r0 + threadIdx.y + i * 8;
    t[threadIdx.y + i * 8][threadIdx.x] = in[(size_t)r * C + c0 + threadIdx.x];
  }
  __syncthreads();
#pragma unroll
  for (int i = 0; i < 4; ++i) {
    const int c = c0 + threadIdx.y + i * 8;
    out[(size_t)c * R + r0 + threadIdx.x] = (bf16)t[threadIdx.x][threadIdx.y + i * 8];
  }
}

// ---------------------------------------------------------------------------
// Kernel 4: fused expert FFN.  grid = (N/16 tiles, E).  8 waves / block.
//   X-tile (16 x 1024 bf16) gathered into LDS via global_load_async_to_lds.
//   For each 128-wide F-chunk:  H = silu(X @ W1T^T) via WMMA -> LDS(bf16),
//   then Y(16x1024) += H @ W2T^T via WMMA (64 persistent acc VGPRs/lane).
//   Finish: + b2, * gate weight, global_atomic_add_f32 scatter.
// ---------------------------------------------------------------------------
static constexpr int MT = 16;    // tokens per tile (= WMMA M)
static constexpr int FC = 128;   // F chunk (= 8 waves * 16 cols)

__global__ __launch_bounds__(256)
void moe_ffn(const bf16* __restrict__ Xb, const bf16* __restrict__ W1T,
             const float* __restrict__ b1, const bf16* __restrict__ W2T,
             const float* __restrict__ b2, const int* __restrict__ counts,
             const int* __restrict__ tok_ids, const float* __restrict__ tok_wts,
             float* __restrict__ out) {
  const int e    = blockIdx.y;
  const int cnt  = counts[e];
  const int base = blockIdx.x * MT;
  if (base >= cnt) return;

  __shared__ bf16  sX[MT][D];      // 32 KB
  __shared__ bf16  sH[MT][FC];     // 4 KB
  __shared__ int   s_tok[MT];
  __shared__ float s_wt[MT];

  const int tid  = threadIdx.x;
  const int lane = tid & 31;
  const int w    = tid >> 5;          // wave 0..7
  const int lr   = lane & 15;
  const int hi   = lane >> 4;         // K-half select
  const int hi16 = hi * 16;

  if (tid < MT) {
    const bool ok = (base + tid) < cnt;
    s_tok[tid] = ok ? tok_ids[e * N + base + tid] : -1;
    s_wt[tid]  = ok ? tok_wts[e * N + base + tid] : 0.f;
  }
  __syncthreads();

  // --- async gather of 16 token rows (bf16) into LDS: 2048 x 16B chunks ---
  {
    const uint64_t xbase = (uint64_t)(uintptr_t)Xb;
#pragma unroll
    for (int i = 0; i < 8; ++i) {
      const int c   = tid + i * 256;          // chunk id
      const int row = c >> 7;                 // 128 chunks per row
      const int off = (c & 127) * 8;          // element offset in row
      int t = s_tok[row]; if (t < 0) t = s_tok[0];
      const uint32_t voff  = (uint32_t)(((size_t)t * D + off) * sizeof(bf16));
      const uint32_t ldsad = (uint32_t)(uintptr_t)&sX[row][off];
      asm volatile("global_load_async_to_lds_b128 %0, %1, %2"
                   :: "v"(ldsad), "v"(voff), "s"(xbase) : "memory");
    }
    asm volatile("s_wait_asynccnt 0" ::: "memory");
  }
  __syncthreads();

  v8f accY[8];
#pragma unroll
  for (int j = 0; j < 8; ++j) accY[j] = {};

  const bf16* W1e = W1T + (size_t)e * F * D;   // [F][D]
  const bf16* W2e = W2T + (size_t)e * D * F;   // [D][F]

  for (int fc = 0; fc < F / FC; ++fc) {
    // ---- phase A: H-chunk = silu(X @ W1) ; one 16-col tile per wave ----
    const int colBase = fc * FC + w * 16;
    {
      v8f acc = {};
      const bf16* Bp = W1e + ((size_t)(colBase + lr)) * D + hi16;
#pragma unroll 8
      for (int kc = 0; kc < D / 32; ++kc) {
        const v16bf a = *(const v16bf*)&sX[lr][kc * 32 + hi16];
        const v16bf b = *(const v16bf*)(Bp + kc * 32);
        acc = wmma_bf16(a, b, acc);
      }
      const float bias = b1[(size_t)e * F + colBase + lr];
#pragma unroll
      for (int r = 0; r < 8; ++r) {
        const int m = r + hi * 8;
        const float v = acc[r] + bias;
        sH[m][w * 16 + lr] = (bf16)(v * fast_sigmoid(v));   // silu
      }
    }
    __syncthreads();

    // ---- phase B: Y += H-chunk @ W2 ; 8 D-tiles per wave ----
#pragma unroll
    for (int kk = 0; kk < FC / 32; ++kk) {
      const v16bf a = *(const v16bf*)&sH[lr][kk * 32 + hi16];
#pragma unroll
      for (int j = 0; j < 8; ++j) {
        const int dcol = w * 128 + j * 16 + lr;
        const v16bf b = *(const v16bf*)(W2e + (size_t)dcol * F +
                                        fc * FC + kk * 32 + hi16);
        accY[j] = wmma_bf16(a, b, accY[j]);
      }
    }
    __syncthreads();
  }

  // ---- epilogue: bias + gate weight, scatter-add into out ----
#pragma unroll
  for (int j = 0; j < 8; ++j) {
    const int dcol = w * 128 + j * 16 + lr;
    const float bias = b2[(size_t)e * D + dcol];
#pragma unroll
    for (int r = 0; r < 8; ++r) {
      const int m = r + hi * 8;
      const int t = s_tok[m];
      if (t >= 0) {
        const float v = (accY[j][r] + bias) * s_wt[m];
        (void)__hip_atomic_fetch_add(&out[(size_t)t * D + dcol], v,
                                     __ATOMIC_RELAXED, __HIP_MEMORY_SCOPE_AGENT);
      }
    }
  }
}

// ---------------------------------------------------------------------------
// Launch
// ---------------------------------------------------------------------------
extern "C" void kernel_launch(void* const* d_in, const int* in_sizes, int n_in,
                              void* d_out, int out_size, void* d_ws, size_t ws_size,
                              hipStream_t stream) {
  (void)in_sizes; (void)n_in; (void)out_size; (void)ws_size;
  const float* x  = (const float*)d_in[0];
  const float* Wg = (const float*)d_in[1];
  const float* bg = (const float*)d_in[2];
  const float* W1 = (const float*)d_in[3];
  const float* b1 = (const float*)d_in[4];
  const float* W2 = (const float*)d_in[5];
  const float* b2 = (const float*)d_in[6];
  float* out = (float*)d_out;

  // workspace carve-up (256B aligned)
  size_t off = 0;
  auto carve = [&](size_t bytes) {
    void* p = (char*)d_ws + off;
    off += (bytes + 255) & ~(size_t)255;
    return p;
  };
  int*   counts  = (int*)  carve(E * sizeof(int));
  int*   tok_ids = (int*)  carve((size_t)E * N * sizeof(int));
  float* tok_wts = (float*)carve((size_t)E * N * sizeof(float));
  bf16*  Xb      = (bf16*) carve((size_t)N * D * sizeof(bf16));
  bf16*  W1T     = (bf16*) carve((size_t)E * F * D * sizeof(bf16));   // [E][F][D]
  bf16*  W2T     = (bf16*) carve((size_t)E * D * F * sizeof(bf16));   // [E][D][F]

  hipMemsetAsync(counts, 0, E * sizeof(int), stream);
  hipMemsetAsync(out, 0, (size_t)N * D * sizeof(float), stream);

  cvt_x_bf16<<<(N * D / 4) / 256, 256, 0, stream>>>(x, Xb);
  // W1 [E][D][F] -> W1T [E][F][D]
  transpose_bf16<<<dim3(F / 32, D / 32, E), dim3(32, 8), 0, stream>>>(W1, W1T, D, F);
  // W2 [E][F][D] -> W2T [E][D][F]
  transpose_bf16<<<dim3(D / 32, F / 32, E), dim3(32, 8), 0, stream>>>(W2, W2T, F, D);

  moe_gate<<<N / 8, 256, 0, stream>>>(x, Wg, bg, counts, tok_ids, tok_wts);
  moe_ffn<<<dim3(N / MT, E), 256, 0, stream>>>(Xb, W1T, b1, W2T, b2,
                                               counts, tok_ids, tok_wts, out);
}
